// LSTMVol_33672543600661
// MI455X (gfx1250) — matrix operand order repaired and checked by reading, not compiled, not run on hardware
//
#include <hip/hip_runtime.h>

// ---------------------------------------------------------------------------
// LSTM (3 layers, B=512, T=256, H=100) on gfx1250 via bf16 WMMA.
// Parallelism: batch rows only (recurrence is independent per row).
// Each wave owns 16 batch rows for the whole time loop; no inter-wave sync.
// ---------------------------------------------------------------------------

typedef __attribute__((ext_vector_type(16))) __bf16      v16bf;
typedef __attribute__((ext_vector_type(8)))  float       v8f;
typedef __attribute__((ext_vector_type(4)))  unsigned int u32x4;

union FragBF {          // 16x32 bf16 operand fragment: 8 VGPRs / lane
    v16bf v;
    u32x4 u[2];
};

#define B_ 512
#define T_ 256
#define H_ 100
#define G_ 400          // 4*H
#define KP 128          // K padded to 128 (4 chunks of 32)

// per-wave LDS: hstage[16][128] bf16 (4096B) + g[16][400] f32 (25600B) + c[16][100] f32 (6400B)
#define WAVE_LDS 36096

__device__ __forceinline__ unsigned short f2bf(float f) {
    unsigned u = __float_as_uint(f);
    u += 0x7FFFu + ((u >> 16) & 1u);          // round-to-nearest-even
    return (unsigned short)(u >> 16);
}
__device__ __forceinline__ float bf2f(unsigned short s) {
    return __uint_as_float(((unsigned)s) << 16);
}
__device__ __forceinline__ float sigmoidf_(float x) {
    return 1.0f / (1.0f + __expf(-x));
}

// ---------------------------------------------------------------------------
// Pack weights: f32 [N][100] -> bf16 [N][128] zero-padded; fuse biases.
// ---------------------------------------------------------------------------
__global__ void lstm_pack_kernel(const float* __restrict__ Wih_rest, // [2][400][100]
                                 const float* __restrict__ Whh,      // [3][400][100]
                                 const float* __restrict__ b_ih,     // [3][400]
                                 const float* __restrict__ b_hh,     // [3][400]
                                 unsigned short* __restrict__ whh_p, // [3][400][128]
                                 unsigned short* __restrict__ wih_p, // [2][400][128]
                                 float* __restrict__ bias_p)         // [3][400]
{
    const int NWHH = 3 * G_ * KP;
    const int NWIH = 2 * G_ * KP;
    int idx = blockIdx.x * blockDim.x + threadIdx.x;
    if (idx < NWHH) {
        int k = idx & (KP - 1);
        int nrow = idx >> 7;                      // l*400+n
        float v = (k < H_) ? Whh[nrow * H_ + k] : 0.0f;
        whh_p[idx] = f2bf(v);
    } else if (idx < NWHH + NWIH) {
        int i2 = idx - NWHH;
        int k = i2 & (KP - 1);
        int nrow = i2 >> 7;
        float v = (k < H_) ? Wih_rest[nrow * H_ + k] : 0.0f;
        wih_p[i2] = f2bf(v);
    } else if (idx < NWHH + NWIH + 3 * G_) {
        int i3 = idx - NWHH - NWIH;
        bias_p[i3] = b_ih[i3] + b_hh[i3];
    }
}

// ---------------------------------------------------------------------------
// Scan kernel: one layer, all T steps. grid=8 blocks x 128 threads (4 waves).
// Wave w of block b owns batch rows [(b*4+w)*16, +16).
// HASPREV=false -> layer 0 (tiny Din=2 input GEMM folded into elementwise).
// HASPREV=true  -> layers 1,2 (input GEMM over prev-layer h via WMMA).
// ---------------------------------------------------------------------------
template <bool HASPREV>
__global__ __launch_bounds__(128, 1)
void lstm_scan_kernel(const float* __restrict__ x,          // [512][256][2] (layer0)
                      const float* __restrict__ Wih0,       // [400][2]      (layer0)
                      const unsigned short* __restrict__ Whh,  // [400][128] bf16
                      const unsigned short* __restrict__ Wih,  // [400][128] bf16 (HASPREV)
                      const float* __restrict__ bias,          // [400]
                      const unsigned short* __restrict__ hs_in,// [T][B][128] bf16 (HASPREV)
                      unsigned short* __restrict__ hs_out)     // [T][B][128] bf16
{
    extern __shared__ __align__(16) char smem[];
    const int lane = threadIdx.x & 31;
    const int wave = threadIdx.x >> 5;
    const int rowbase = (blockIdx.x * 4 + wave) * 16;

    char* wbase = smem + wave * WAVE_LDS;
    unsigned short* hstage = (unsigned short*)wbase;            // [16][128] bf16
    float* gbuf = (float*)(wbase + 4096);                       // [16][400]
    float* cbuf = (float*)(wbase + 4096 + 25600);               // [16][100]

    // h = c = 0 (padding cols of hstage stay zero forever)
    for (int i = lane; i < 16 * KP; i += 32) hstage[i] = 0;
    for (int i = lane; i < 16 * H_; i += 32) cbuf[i] = 0.0f;

    const int lhi = lane >> 4;   // 0: K 0-7,16-23 of chunk, 1: K 8-15,24-31
    const int llo = lane & 15;

    for (int t = 0; t < T_; ++t) {
        // ---- A fragments for current h (from LDS hstage, row-major bf16) ----
        FragBF Ah[4];
#pragma unroll
        for (int kc = 0; kc < 4; ++kc) {
            const char* p = (const char*)hstage + llo * (KP * 2) + lhi * 16 + kc * 64;
            Ah[kc].u[0] = *(const u32x4*)(p);
            Ah[kc].u[1] = *(const u32x4*)(p + 32);
        }
        // ---- A fragments for prev-layer h at step t (global, L2-resident) ----
        FragBF Ap[4];
        if constexpr (HASPREV) {
            const char* pb = (const char*)hs_in +
                ((size_t)(t * B_ + rowbase + llo) * KP) * 2 + lhi * 16;
#pragma unroll
            for (int kc = 0; kc < 4; ++kc) {
                Ap[kc].u[0] = *(const u32x4*)(pb + kc * 64);
                Ap[kc].u[1] = *(const u32x4*)(pb + kc * 64 + 32);
            }
        }

        // ---- g = h @ Whh^T (+ h_prev @ Wih^T), 25 N-tiles of 16 ----
        for (int nt = 0; nt < 25; ++nt) {
            v8f acc = {};
            const int nrow = nt * 16 + llo;
            {
                const char* wb = (const char*)Whh + ((size_t)nrow * KP) * 2 + lhi * 16;
#pragma unroll
                for (int kc = 0; kc < 4; ++kc) {
                    FragBF Bm;
                    Bm.u[0] = *(const u32x4*)(wb + kc * 64);
                    Bm.u[1] = *(const u32x4*)(wb + kc * 64 + 32);
                    acc = __builtin_amdgcn_wmma_f32_16x16x32_bf16(
                        false, Ah[kc].v, false, Bm.v, (short)0, acc, false, false);
                }
            }
            if constexpr (HASPREV) {
                const char* wb2 = (const char*)Wih + ((size_t)nrow * KP) * 2 + lhi * 16;
#pragma unroll
                for (int kc = 0; kc < 4; ++kc) {
                    FragBF Bm;
                    Bm.u[0] = *(const u32x4*)(wb2 + kc * 64);
                    Bm.u[1] = *(const u32x4*)(wb2 + kc * 64 + 32);
                    acc = __builtin_amdgcn_wmma_f32_16x16x32_bf16(
                        false, Ap[kc].v, false, Bm.v, (short)0, acc, false, false);
                }
            }
            // C/D layout: lane l, vgpr v -> (M = v + 8*(l>>4), N = nt*16 + (l&15))
#pragma unroll
            for (int v = 0; v < 8; ++v) {
                int m = v + 8 * lhi;
                gbuf[m * G_ + nt * 16 + llo] = acc[v];
            }
        }

        // ---- elementwise gate math: 16 rows x 100 cells = 1600 items / wave ----
        for (int it = 0; it < 50; ++it) {
            int flat = it * 32 + lane;           // 0..1599
            int m = flat / H_;
            int j = flat - m * H_;
            float gi = gbuf[m * G_ + j];
            float gf = gbuf[m * G_ + H_ + j];
            float gg = gbuf[m * G_ + 2 * H_ + j];
            float go = gbuf[m * G_ + 3 * H_ + j];
            if constexpr (!HASPREV) {
                const float* xp = x + ((size_t)(rowbase + m) * T_ + t) * 2;
                float x0 = xp[0], x1 = xp[1];
                gi += x0 * Wih0[(j)          * 2] + x1 * Wih0[(j)          * 2 + 1] + bias[j];
                gf += x0 * Wih0[(H_ + j)     * 2] + x1 * Wih0[(H_ + j)     * 2 + 1] + bias[H_ + j];
                gg += x0 * Wih0[(2 * H_ + j) * 2] + x1 * Wih0[(2 * H_ + j) * 2 + 1] + bias[2 * H_ + j];
                go += x0 * Wih0[(3 * H_ + j) * 2] + x1 * Wih0[(3 * H_ + j) * 2 + 1] + bias[3 * H_ + j];
            } else {
                gi += bias[j];
                gf += bias[H_ + j];
                gg += bias[2 * H_ + j];
                go += bias[3 * H_ + j];
            }
            float c  = cbuf[m * H_ + j];
            float iv = sigmoidf_(gi);
            float fv = sigmoidf_(gf);
            float ov = sigmoidf_(go);
            float cn = fv * c + iv * tanhf(gg);
            float hn = ov * tanhf(cn);
            cbuf[m * H_ + j] = cn;
            unsigned short hb = f2bf(hn);
            hstage[m * KP + j] = hb;
            hs_out[((size_t)t * B_ + rowbase + m) * KP + j] = hb;
        }
        // same-wave DS ordering guarantees the RAW into next step's A loads
    }
}

// ---------------------------------------------------------------------------
// Head: out[b] = head_b + sum_j h_L2[T-1][b][j] * head_w[j]
// ---------------------------------------------------------------------------
__global__ void lstm_head_kernel(const unsigned short* __restrict__ hs, // [T][B][128]
                                 const float* __restrict__ head_w,      // [100]
                                 const float* __restrict__ head_b,      // [1]
                                 float* __restrict__ out)               // [512]
{
    int b = blockIdx.x * blockDim.x + threadIdx.x;
    if (b < B_) {
        const unsigned short* hp = hs + ((size_t)((T_ - 1) * B_ + b)) * KP;
        float s = head_b[0];
#pragma unroll 4
        for (int j = 0; j < H_; ++j) s += bf2f(hp[j]) * head_w[j];
        out[b] = s;
    }
}

// ---------------------------------------------------------------------------
// Workspace layout (bytes):
//   [0,307200)            whh_p  bf16 [3][400][128]
//   [307200,512000)       wih_p  bf16 [2][400][128]
//   [512000,516800)       bias_p f32  [3][400]
//   [524288, +33554432)   hsA    bf16 [256][512][128]
//   [.. , +33554432)      hsB    bf16 [256][512][128]
// total ~64.5 MB
// ---------------------------------------------------------------------------
extern "C" void kernel_launch(void* const* d_in, const int* in_sizes, int n_in,
                              void* d_out, int out_size, void* d_ws, size_t ws_size,
                              hipStream_t stream) {
    const float* x        = (const float*)d_in[0];
    const float* Wih0     = (const float*)d_in[1];
    const float* Wih_rest = (const float*)d_in[2];
    const float* Whh      = (const float*)d_in[3];
    const float* b_ih     = (const float*)d_in[4];
    const float* b_hh     = (const float*)d_in[5];
    const float* head_w   = (const float*)d_in[6];
    const float* head_b   = (const float*)d_in[7];
    float* out = (float*)d_out;

    char* ws = (char*)d_ws;
    unsigned short* whh_p = (unsigned short*)(ws);
    unsigned short* wih_p = (unsigned short*)(ws + 307200);
    float*          bias_p = (float*)(ws + 512000);
    const size_t HS_BYTES = (size_t)T_ * B_ * KP * 2;           // 33,554,432
    unsigned short* hsA = (unsigned short*)(ws + 524288);
    unsigned short* hsB = (unsigned short*)(ws + 524288 + HS_BYTES);

    // zero hidden buffers once per launch (padding cols must read as 0)
    hipMemsetAsync(ws + 524288, 0, 2 * HS_BYTES, stream);

    // pack weights to zero-padded bf16, fuse biases
    {
        int total = 3 * G_ * KP + 2 * G_ * KP + 3 * G_;
        lstm_pack_kernel<<<(total + 255) / 256, 256, 0, stream>>>(
            Wih_rest, Whh, b_ih, b_hh, whh_p, wih_p, bias_p);
    }

    const size_t lds = 4 * (size_t)WAVE_LDS;   // 144,384 B (4 waves/block)

    // layer 0: x -> hsA
    lstm_scan_kernel<false><<<8, 128, lds, stream>>>(
        x, Wih0, whh_p, nullptr, bias_p, nullptr, hsA);
    // layer 1: hsA -> hsB
    lstm_scan_kernel<true><<<8, 128, lds, stream>>>(
        nullptr, nullptr, whh_p + 1 * G_ * KP, wih_p, bias_p + G_, hsA, hsB);
    // layer 2: hsB -> hsA
    lstm_scan_kernel<true><<<8, 128, lds, stream>>>(
        nullptr, nullptr, whh_p + 2 * G_ * KP, wih_p + G_ * KP, bias_p + 2 * G_, hsB, hsA);

    // head
    lstm_head_kernel<<<(B_ + 255) / 256, 256, 0, stream>>>(hsA, head_w, head_b, out);
}
